// LFM3BAttention_63531156242849
// MI455X (gfx1250) — matrix operand order repaired and checked by reading, not compiled, not run on hardware
//
#include <hip/hip_runtime.h>
#include <hip/hip_bf16.h>

#define S   2048
#define HID 2048
#define NH  32
#define NKV 8
#define HD  64
#define KVD (NKV*HD)   // 512

typedef __attribute__((ext_vector_type(16))) __bf16          v16bf;
typedef __attribute__((ext_vector_type(8)))  float           v8f;
typedef __attribute__((ext_vector_type(8)))  unsigned short  v8u;
typedef __attribute__((ext_vector_type(16))) unsigned short  v16u;

#if defined(__AMDGCN__) && __has_builtin(__builtin_amdgcn_tensor_load_to_lds) && __has_builtin(__builtin_amdgcn_s_wait_tensorcnt)
#define USE_TDM 1
#else
#define USE_TDM 0
#endif

__device__ __forceinline__ unsigned short f2bf(float f) {
  union { float f; unsigned u; } c; c.f = f;
  unsigned r = (c.u + 0x7FFFu + ((c.u >> 16) & 1u)) >> 16;   // RNE
  return (unsigned short)r;
}
__device__ __forceinline__ float bf2f(unsigned short h) {
  union { unsigned u; float f; } c; c.u = ((unsigned)h) << 16;
  return c.f;
}
__device__ __forceinline__ v16u cat8(v8u lo, v8u hi) {
  return __builtin_shufflevector(lo, hi, 0,1,2,3,4,5,6,7,8,9,10,11,12,13,14,15);
}
__device__ __forceinline__ v16bf as_bf(v16u x) {
  union { v16u u; v16bf b; } c; c.u = x; return c.b;
}
__device__ __forceinline__ v8f wmma_bf(v16u a, v16u b, v8f c) {
  return __builtin_amdgcn_wmma_f32_16x16x32_bf16(
      /*neg_a=*/false, as_bf(a), /*neg_b=*/false, as_bf(b),
      /*c_mod=*/(short)0, c, /*reuse_a=*/false, /*reuse_b=*/false);
}
__device__ __forceinline__ v8f vzero8() {
  v8f z;
#pragma unroll
  for (int i = 0; i < 8; ++i) z[i] = 0.f;
  return z;
}

#if USE_TDM
typedef __attribute__((ext_vector_type(4))) unsigned int u32x4;
typedef __attribute__((ext_vector_type(8))) int          i32x8;
typedef __attribute__((ext_vector_type(4))) int          i32x4;

// TDM: pull a 128-row x 64-col bf16 tile (row stride = strideElems) into LDS,
// with the mover inserting 8 shorts of padding per row -> LDS row stride 72.
__device__ __forceinline__ void tdm_load_tile(const unsigned short* lds_dst,
                                              const unsigned short* gsrc,
                                              int strideElems) {
  unsigned long long ga = (unsigned long long)(uintptr_t)gsrc;
  unsigned ldsOff = (unsigned)(uintptr_t)lds_dst;  // low 32b of flat = LDS offset
  u32x4 g0;
  g0[0] = 1u;                                      // count=1, user-mode load
  g0[1] = ldsOff;                                  // lds_addr
  g0[2] = (unsigned)ga;                            // global_addr[31:0]
  g0[3] = (unsigned)((ga >> 32) & 0x01FFFFFFu) | (2u << 30);  // addr[56:32]|type=2
  i32x8 g1;
  unsigned long long st = (unsigned long long)strideElems;
  g1[0] = (int)((1u << 16) |                       // data_size = 2 bytes
                (1u << 20) |                       // pad_enable
                (4u << 22) |                       // pad_interval: 32 DW (=64 bf16 row)
                (3u << 25));                       // pad_amount: 4 DW (=8 shorts)
  g1[1] = (int)(64u << 16);                        // tensor_dim0 = 64 (lo16)
  g1[2] = (int)(128u << 16);                       // dim0 hi=0 | tensor_dim1=128 (lo16)
  g1[3] = (int)(64u << 16);                        // dim1 hi=0 | tile_dim0 = 64
  g1[4] = (int)(128u);                             // tile_dim1 = 128 | tile_dim2 = 0
  g1[5] = (int)(st & 0xFFFFFFFFull);               // tensor_dim0_stride lo32
  g1[6] = (int)((st >> 32) & 0xFFFFull);           // stride hi16 | dim1_stride lo = 0
  g1[7] = 0;
  i32x4 zero4 = {0, 0, 0, 0};
  i32x8 zero8 = {0, 0, 0, 0, 0, 0, 0, 0};
  // clang-23 / therock-10.0 form: 6 args (g0, g1, g2, g3, extra, cpol)
  __builtin_amdgcn_tensor_load_to_lds(g0, g1, zero4, zero4, zero8, 0);
}
#endif

// ---------------- elementwise prep kernels ----------------

__global__ __launch_bounds__(256) void cast_bf16_kernel(
    const float* __restrict__ in, unsigned short* __restrict__ out, int n) {
  int i = blockIdx.x * 256 + threadIdx.x;
  if (i < n) out[i] = f2bf(in[i]);
}

// in: fp32 [Kd][Nd] row-major  ->  out: bf16 [Nd][Kd] (i.e. B^T)
__global__ __launch_bounds__(256) void transpose_cast_kernel(
    const float* __restrict__ in, unsigned short* __restrict__ out, int Kd, int Nd) {
  int i = blockIdx.x * 256 + threadIdx.x;
  if (i < Kd * Nd) {
    int k = i / Nd, n = i % Nd;
    out[(size_t)n * Kd + k] = f2bf(in[i]);
  }
}

// in-place RoPE on bf16 [S][nheads*64]; thread handles pair (d, d+32)
__global__ __launch_bounds__(256) void rope_kernel(
    unsigned short* __restrict__ X, int nheads) {
  int i = blockIdx.x * 256 + threadIdx.x;
  int total = S * nheads * 32;
  if (i >= total) return;
  int d = i & 31; int t = i >> 5; int hh = t % nheads; int s = t / nheads;
  unsigned short* p = X + (size_t)s * nheads * HD + hh * HD + d;
  float x1 = bf2f(p[0]), x2 = bf2f(p[32]);
  // theta = s * 10000^(-d/32) = s * exp(-d * ln(10000)/32)
  float th = (float)s * __expf(-(float)d * 0.28782313662425572f);
  float sn, cs; __sincosf(th, &sn, &cs);
  p[0]  = f2bf(x1 * cs - x2 * sn);
  p[32] = f2bf(x2 * cs + x1 * sn);
}

// V bf16 [S][KVD] -> Vt bf16 [KVD][S]
__global__ __launch_bounds__(256) void transpose_v_kernel(
    const unsigned short* __restrict__ V, unsigned short* __restrict__ Vt) {
  int i = blockIdx.x * 256 + threadIdx.x;  // S*KVD exact
  int s = i / KVD, c = i % KVD;
  Vt[(size_t)c * S + s] = V[i];
}

// ---------------- tiled bf16 WMMA GEMM, TDM-fed, double-buffered ----------------
// C[M][N] (bf16 or fp32) = A[M][K](bf16 row-major) * Bt[N][K](bf16 = B^T)
// 256 thr (8 waves). Tile 128x128, BK=64. Wave tile 64x32 -> 16 WMMA / K-tile.
template <bool OUT_BF16>
__global__ __launch_bounds__(256) void gemm_bf16_kernel(
    const unsigned short* __restrict__ A,
    const unsigned short* __restrict__ Bt,
    void* __restrict__ Cptr, int M, int N, int K) {
#if USE_TDM
  __shared__ unsigned short sA[2][128 * 72];   // padded stride 72 (144B)
  __shared__ unsigned short sB[2][128 * 72];
#else
  __shared__ unsigned short sA[1][128 * 72];
  __shared__ unsigned short sB[1][128 * 72];
#endif
  const int tid = threadIdx.x;
  const int wid = tid >> 5, lane = tid & 31;
  const int lh  = lane >> 4, l16 = lane & 15;
  const int m0  = blockIdx.y * 128, n0 = blockIdx.x * 128;
  const int wm  = (wid >> 2) * 64;             // 0 / 64
  const int wn  = (wid & 3) * 32;              // 0..96

  v8f acc[8];
#pragma unroll
  for (int i = 0; i < 8; ++i) acc[i] = vzero8();

  const int nk = K >> 6;                       // K-tiles of 64

#if USE_TDM
  if (wid == 0) {                              // prologue: fetch tile 0
    tdm_load_tile(&sA[0][0], A  + (size_t)m0 * K, K);
    tdm_load_tile(&sB[0][0], Bt + (size_t)n0 * K, K);
  }
#endif

  for (int kt = 0; kt < nk; ++kt) {
#if USE_TDM
    const int cur = kt & 1;
    if (wid == 0) __builtin_amdgcn_s_wait_tensorcnt(0);  // tile kt landed
    __syncthreads();                                     // visible; prev buf free
    if (wid == 0 && kt + 1 < nk) {                       // prefetch tile kt+1
      const int k1 = (kt + 1) << 6;
      tdm_load_tile(&sA[cur ^ 1][0], A  + (size_t)m0 * K + k1, K);
      tdm_load_tile(&sB[cur ^ 1][0], Bt + (size_t)n0 * K + k1, K);
    }
    const unsigned short* pa = sA[cur];
    const unsigned short* pb = sB[cur];
#else
    __syncthreads();
    {
      const int r = tid >> 1, c = (tid & 1) * 32;
      const unsigned short* ga = A  + (size_t)(m0 + r) * K + (kt << 6) + c;
      const unsigned short* gb = Bt + (size_t)(n0 + r) * K + (kt << 6) + c;
      unsigned short* la = &sA[0][r * 72 + c];
      unsigned short* lb = &sB[0][r * 72 + c];
      v8u a0 = *(const v8u*)ga, a1 = *(const v8u*)(ga + 8);
      v8u a2 = *(const v8u*)(ga + 16), a3 = *(const v8u*)(ga + 24);
      v8u b0 = *(const v8u*)gb, b1 = *(const v8u*)(gb + 8);
      v8u b2 = *(const v8u*)(gb + 16), b3 = *(const v8u*)(gb + 24);
      *(v8u*)la = a0; *(v8u*)(la + 8) = a1; *(v8u*)(la + 16) = a2; *(v8u*)(la + 24) = a3;
      *(v8u*)lb = b0; *(v8u*)(lb + 8) = b1; *(v8u*)(lb + 16) = b2; *(v8u*)(lb + 24) = b3;
    }
    __syncthreads();
    const unsigned short* pa = sA[0];
    const unsigned short* pb = sB[0];
#endif

#pragma unroll
    for (int ks = 0; ks < 2; ++ks) {           // two 16x16x32 K-substeps
      v16u af[4], bfr[2];
#pragma unroll
      for (int mi = 0; mi < 4; ++mi) {
        // A-frag: lanes 0-15 rows, K chunks {0..7,16..23}; lanes 16-31 {8..15,24..31}
        const unsigned short* p = pa + (wm + mi * 16 + l16) * 72 + ks * 32 + lh * 8;
        af[mi] = cat8(*(const v8u*)p, *(const v8u*)(p + 16));
      }
#pragma unroll
      for (int ni = 0; ni < 2; ++ni) {
        // B-frag: lane = column, K chunks {0..7,16..23}; lanes 16-31 {8..15,24..31}
        const unsigned short* p = pb + (wn + ni * 16 + l16) * 72 + ks * 32 + lh * 8;
        bfr[ni] = cat8(*(const v8u*)p, *(const v8u*)(p + 16));
      }
#pragma unroll
      for (int mi = 0; mi < 4; ++mi)
#pragma unroll
        for (int ni = 0; ni < 2; ++ni)
          acc[mi * 2 + ni] = wmma_bf(af[mi], bfr[ni], acc[mi * 2 + ni]);
    }
  }

#pragma unroll
  for (int mi = 0; mi < 4; ++mi)
#pragma unroll
    for (int ni = 0; ni < 2; ++ni)
#pragma unroll
      for (int r = 0; r < 8; ++r) {
        int row = m0 + wm + mi * 16 + lh * 8 + r;   // C layout: vgpr r, half=+8
        int col = n0 + wn + ni * 16 + l16;
        float v = acc[mi * 2 + ni][r];
        if (OUT_BF16) ((unsigned short*)Cptr)[(size_t)row * N + col] = f2bf(v);
        else          ((float*)Cptr)[(size_t)row * N + col] = v;
      }
}

// ---------------- fused flash attention (bf16 WMMA, fp32 softmax) ----------------
// Grid: (S/128, NH). 8 waves/WG, each wave owns 16 queries of head blockIdx.y.
__global__ __launch_bounds__(256) void attn_kernel(
    const unsigned short* __restrict__ Q,    // [S][HID]  (post-RoPE)
    const unsigned short* __restrict__ Kc,   // [S][KVD]  (post-RoPE)
    const unsigned short* __restrict__ Vt,   // [KVD][S]
    unsigned short* __restrict__ O) {        // [S][HID]
  __shared__ unsigned short sP[8][16 * 72];  // per-wave P tile, padded stride 72
  const int tid = threadIdx.x;
  const int wid = tid >> 5, lane = tid & 31;
  const int lh  = lane >> 4, l16 = lane & 15;
  const int h   = blockIdx.y;
  const int kv  = h >> 2;                    // GQA: 4 query heads per kv head
  const int q0  = blockIdx.x * 128 + wid * 16;

  v16u qf[2];
  {
    const unsigned short* qp = Q + (size_t)(q0 + l16) * HID + h * HD;
#pragma unroll
    for (int f = 0; f < 2; ++f) {
      const unsigned short* p = qp + f * 32 + lh * 8;
      qf[f] = cat8(*(const v8u*)p, *(const v8u*)(p + 16));
    }
  }

  float mrow[8], lrowv[8];
#pragma unroll
  for (int r = 0; r < 8; ++r) { mrow[r] = -3.0e38f; lrowv[r] = 0.f; }
  v8f acc[4];
#pragma unroll
  for (int i = 0; i < 4; ++i) acc[i] = vzero8();

  const int nkb = (q0 + 15) / 64 + 1;        // causal: key blocks of 64
  for (int kb = 0; kb < nkb; ++kb) {
    const int kbase = kb * 64;

    // scores = Q K^T over D=64 (two 16x16x32 steps), 4 key sub-tiles
    v8f s[4];
#pragma unroll
    for (int n = 0; n < 4; ++n) {
      v8f sc = vzero8();
#pragma unroll
      for (int f = 0; f < 2; ++f) {
        const unsigned short* kp =
            Kc + (size_t)(kbase + n * 16 + l16) * KVD + kv * HD + f * 32 + lh * 8;
        v16u kf = cat8(*(const v8u*)kp, *(const v8u*)(kp + 16));
        sc = wmma_bf(qf[f], kf, sc);
      }
      s[n] = sc;
    }

    // scale 1/sqrt(64) + causal mask
#pragma unroll
    for (int n = 0; n < 4; ++n) {
      int key = kbase + n * 16 + l16;
#pragma unroll
      for (int r = 0; r < 8; ++r) {
        int qr = q0 + lh * 8 + r;
        float v = s[n][r] * 0.125f;
        s[n][r] = (key > qr) ? -1.0e9f : v;
      }
    }

    // online softmax: row reductions across the 16-lane column groups
    float rmax[8];
#pragma unroll
    for (int r = 0; r < 8; ++r)
      rmax[r] = fmaxf(fmaxf(s[0][r], s[1][r]), fmaxf(s[2][r], s[3][r]));
    for (int off = 8; off >= 1; off >>= 1)
#pragma unroll
      for (int r = 0; r < 8; ++r)
        rmax[r] = fmaxf(rmax[r], __shfl_xor(rmax[r], off, 16));

    float rsum[8];
#pragma unroll
    for (int r = 0; r < 8; ++r) {
      float mn = fmaxf(mrow[r], rmax[r]);
      float sc = __expf(mrow[r] - mn);
      mrow[r] = mn;
      lrowv[r] *= sc;
#pragma unroll
      for (int dn = 0; dn < 4; ++dn) acc[dn][r] *= sc;
      rsum[r] = 0.f;
    }

    // P = exp(S - m): C-layout -> LDS -> reread as A-layout fragments
    unsigned short* pw = sP[wid];
#pragma unroll
    for (int n = 0; n < 4; ++n)
#pragma unroll
      for (int r = 0; r < 8; ++r) {
        float p = __expf(s[n][r] - mrow[r]);
        rsum[r] += p;
        pw[(lh * 8 + r) * 72 + n * 16 + l16] = f2bf(p);
      }
    for (int off = 8; off >= 1; off >>= 1)
#pragma unroll
      for (int r = 0; r < 8; ++r)
        rsum[r] += __shfl_xor(rsum[r], off, 16);
#pragma unroll
    for (int r = 0; r < 8; ++r) lrowv[r] += rsum[r];

    v16u pf[2];
#pragma unroll
    for (int f = 0; f < 2; ++f) {
      const unsigned short* p = pw + l16 * 72 + f * 32 + lh * 8;
      pf[f] = cat8(*(const v8u*)p, *(const v8u*)(p + 16));
    }

    // acc += P * V  (V^T layout -> contiguous B-frags)
#pragma unroll
    for (int dn = 0; dn < 4; ++dn)
#pragma unroll
      for (int f = 0; f < 2; ++f) {
        const unsigned short* vp =
            Vt + ((size_t)kv * HD + dn * 16 + l16) * S + kbase + f * 32 + lh * 8;
        v16u vf = cat8(*(const v8u*)vp, *(const v8u*)(vp + 16));
        acc[dn] = wmma_bf(pf[f], vf, acc[dn]);
      }
  }

#pragma unroll
  for (int dn = 0; dn < 4; ++dn)
#pragma unroll
    for (int r = 0; r < 8; ++r) {
      int row = q0 + lh * 8 + r;
      int col = h * HD + dn * 16 + l16;
      O[(size_t)row * HID + col] = f2bf(acc[dn][r] / lrowv[r]);
    }
}

// ---------------- launch ----------------

extern "C" void kernel_launch(void* const* d_in, const int* in_sizes, int n_in,
                              void* d_out, int out_size, void* d_ws, size_t ws_size,
                              hipStream_t stream) {
  (void)in_sizes; (void)n_in; (void)out_size; (void)ws_size;
  const float* hs = (const float*)d_in[0];
  const float* Wq = (const float*)d_in[1];
  const float* Wk = (const float*)d_in[2];
  const float* Wv = (const float*)d_in[3];
  const float* Wo = (const float*)d_in[4];
  // d_in[5] = attention_mask: causal, applied analytically — never read.

  char* w = (char*)d_ws;
  size_t off = 0;
  auto take = [&](size_t elems) -> unsigned short* {
    unsigned short* p = (unsigned short*)(w + off);
    off += (elems * sizeof(unsigned short) + 255) & ~(size_t)255;
    return p;
  };
  unsigned short* hsb = take((size_t)S * HID);    // hidden, bf16
  unsigned short* WqT = take((size_t)HID * HID);  // Wq^T bf16
  unsigned short* WkT = take((size_t)KVD * HID);
  unsigned short* WvT = take((size_t)KVD * HID);
  unsigned short* WoT = take((size_t)HID * HID);
  unsigned short* Qb  = take((size_t)S * HID);
  unsigned short* Kb  = take((size_t)S * KVD);
  unsigned short* Vb  = take((size_t)S * KVD);
  unsigned short* Vt  = take((size_t)KVD * S);
  unsigned short* Ab  = take((size_t)S * HID);    // attention out, bf16

  dim3 blk(256);
  cast_bf16_kernel<<<(S * HID + 255) / 256, blk, 0, stream>>>(hs, hsb, S * HID);
  transpose_cast_kernel<<<(HID * HID + 255) / 256, blk, 0, stream>>>(Wq, WqT, HID, HID);
  transpose_cast_kernel<<<(HID * KVD + 255) / 256, blk, 0, stream>>>(Wk, WkT, HID, KVD);
  transpose_cast_kernel<<<(HID * KVD + 255) / 256, blk, 0, stream>>>(Wv, WvT, HID, KVD);
  transpose_cast_kernel<<<(HID * HID + 255) / 256, blk, 0, stream>>>(Wo, WoT, HID, HID);

  gemm_bf16_kernel<true><<<dim3(HID / 128, S / 128), blk, 0, stream>>>(hsb, WqT, Qb, S, HID, HID);
  gemm_bf16_kernel<true><<<dim3(KVD / 128, S / 128), blk, 0, stream>>>(hsb, WkT, Kb, S, KVD, HID);
  gemm_bf16_kernel<true><<<dim3(KVD / 128, S / 128), blk, 0, stream>>>(hsb, WvT, Vb, S, KVD, HID);

  rope_kernel<<<(S * NH * 32 + 255) / 256, blk, 0, stream>>>(Qb, NH);
  rope_kernel<<<(S * NKV * 32 + 255) / 256, blk, 0, stream>>>(Kb, NKV);
  transpose_v_kernel<<<(S * KVD + 255) / 256, blk, 0, stream>>>(Vb, Vt);

  attn_kernel<<<dim3(S / 128, NH), blk, 0, stream>>>(Qb, Kb, Vt, Ab);

  gemm_bf16_kernel<false><<<dim3(HID / 128, S / 128), blk, 0, stream>>>(Ab, WoT, d_out, S, HID, HID);
}
